// MultiHeadedAttention_43284680409335
// MI455X (gfx1250) — matrix operand order repaired and checked by reading, not compiled, not run on hardware
//
#include <hip/hip_runtime.h>

typedef unsigned short u16;
typedef __attribute__((ext_vector_type(16))) __bf16 v16bf;
typedef __attribute__((ext_vector_type(8)))  float  v8f;
typedef __attribute__((ext_vector_type(4)))  unsigned int v4u;
typedef __attribute__((ext_vector_type(8)))  int v8i;
typedef __attribute__((ext_vector_type(4)))  int v4i;

#define BATCH   32
#define D_MODEL 256
#define NPTS    512
#define HEADS   4
#define HDIM    64   // D_MODEL / HEADS

#if defined(__has_builtin)
#if __has_builtin(__builtin_amdgcn_tensor_load_to_lds) && __has_builtin(__builtin_amdgcn_s_wait_tensorcnt)
#define USE_TDM 1
#endif
#endif
#ifndef USE_TDM
#define USE_TDM 0
#endif

union Frag {
    v16bf v;
    uint4 q[2];
};

__device__ __forceinline__ u16 f2bf(float f) {
    unsigned u = __float_as_uint(f);
    u += 0x7FFFu + ((u >> 16) & 1u);   // round to nearest even
    return (u16)(u >> 16);
}

// ---------------------------------------------------------------------------
// TDM: 2D bf16 tile load (global -> LDS).  tile_w elems per row, tile_h rows,
// global row stride in elems.  Packs D# group0/group1 per CDNA5 ISA 8.3/8.4.
// ---------------------------------------------------------------------------
#if USE_TDM
__device__ __forceinline__ void tdm_load_2d_bf16(unsigned lds_off, const void* gptr,
                                                 unsigned tile_w, unsigned tile_h,
                                                 unsigned row_stride) {
    unsigned long long ga = (unsigned long long)gptr;
    v4u g0;
    g0[0] = 1u;                                            // count=1 (valid user D#)
    g0[1] = lds_off;                                       // lds_addr (bytes)
    g0[2] = (unsigned)(ga & 0xFFFFFFFFull);                // global_addr[31:0]
    g0[3] = (unsigned)((ga >> 32) & 0x01FFFFFFull) | (2u << 30);  // addr[56:32] | type=2
    v8i g1;
    g1[0] = (int)(1u << 16);                               // data_size=1 (2B), wg_mask=0
    g1[1] = (int)(tile_w << 16);                           // tensor_dim0[15:0] @ [63:48]
    g1[2] = (int)((tile_w >> 16) | (tile_h << 16));        // tdim0[31:16] | tdim1[15:0]
    g1[3] = (int)((tile_h >> 16) | (tile_w << 16));        // tdim1[31:16] | tile_dim0
    g1[4] = (int)(tile_h & 0xFFFFu);                       // tile_dim1 | tile_dim2=0
    g1[5] = (int)row_stride;                               // tensor_dim0_stride[31:0]
    g1[6] = 0;                                             // stride hi | dim1_stride lo
    g1[7] = 0;
    v4i z4 = {0, 0, 0, 0};
#if __clang_major__ >= 23
    v8i z8 = {0, 0, 0, 0, 0, 0, 0, 0};
    __builtin_amdgcn_tensor_load_to_lds(g0, g1, z4, z4, z8, 0);
#else
    __builtin_amdgcn_tensor_load_to_lds(g0, g1, z4, z4, 0);
#endif
}
#endif

// ---------------------------------------------------------------------------
// f32 -> bf16 elementwise convert (weights)
// ---------------------------------------------------------------------------
__global__ __launch_bounds__(256) void cvt_bf16_kernel(const float* __restrict__ in,
                                                       u16* __restrict__ out, int count) {
    int i = blockIdx.x * 256 + threadIdx.x;
    if (i < count) out[i] = f2bf(in[i]);
}

// ---------------------------------------------------------------------------
// f32 [B,D,N] -> bf16 [B,N,D] transpose-convert (32x32 LDS tile)
// ---------------------------------------------------------------------------
__global__ __launch_bounds__(256) void cvt_tr_kernel(const float* __restrict__ in,
                                                     u16* __restrict__ out) {
    __shared__ float t[32][33];
    int b  = blockIdx.z;
    int n0 = blockIdx.x * 32;
    int d0 = blockIdx.y * 32;
    int tid = threadIdx.x;
    int r  = tid >> 3;         // 0..31
    int c4 = (tid & 7) * 4;    // 0,4,...,28

    float4 v = *(const float4*)(in + ((size_t)b * D_MODEL + d0 + r) * NPTS + n0 + c4);
    t[r][c4 + 0] = v.x; t[r][c4 + 1] = v.y; t[r][c4 + 2] = v.z; t[r][c4 + 3] = v.w;
    __syncthreads();

    unsigned p0 = (unsigned)f2bf(t[c4 + 0][r]) | ((unsigned)f2bf(t[c4 + 1][r]) << 16);
    unsigned p1 = (unsigned)f2bf(t[c4 + 2][r]) | ((unsigned)f2bf(t[c4 + 3][r]) << 16);
    uint2 pk = make_uint2(p0, p1);
    *(uint2*)(out + ((size_t)b * NPTS + n0 + r) * D_MODEL + d0 + c4) = pk;
}

// ---------------------------------------------------------------------------
// dist modulation: dm[b,n,m] = proj_dist[n, rank(m)] * dist(b,n,m) / sqrt(HD)
// ---------------------------------------------------------------------------
__global__ __launch_bounds__(512) void dist_kernel(const float* __restrict__ src,
                                                   const float* __restrict__ dst,
                                                   const float* __restrict__ proj,
                                                   float* __restrict__ dm) {
    int n = blockIdx.x;
    int b = blockIdx.y;
    int m = threadIdx.x;
    __shared__ float drow[NPTS];

    float sx = src[((size_t)b * NPTS + n) * 2 + 0];
    float sy = src[((size_t)b * NPTS + n) * 2 + 1];
    float dx = sx - dst[((size_t)b * NPTS + m) * 2 + 0];
    float dy = sy - dst[((size_t)b * NPTS + m) * 2 + 1];
    float d = sqrtf(dx * dx + dy * dy);
    drow[m] = d;
    __syncthreads();

    int rank = 0;
    #pragma unroll 4
    for (int j = 0; j < NPTS; ++j) {
        float dj = drow[j];
        rank += (dj < d) || (dj == d && j < m);
    }
    dm[((size_t)b * NPTS + n) * NPTS + m] = proj[(size_t)n * NPTS + rank] * d * 0.125f;
}

// ---------------------------------------------------------------------------
// Projection GEMM, LDS-free: out = W[D,D] x X (+bias).
// W: bf16 row-major [dout][din].  Xt: bf16 [b][n][din] (din contiguous).
// Block 256 (8 waves) computes 64(dout) x 64(n); each wave a 16x32 tile.
// mode 0: q/k layout  out[((b*H+h)*N+n)*HD+hd]  (bf16)
// mode 1: v   layout  out[((b*H+h)*HD+hd)*N+n]  (bf16)
// mode 2: final       out[(b*D+d)*N+n]          (f32, = d_out)
// ---------------------------------------------------------------------------
__global__ __launch_bounds__(256) void proj_gemm_kernel(const u16* __restrict__ W,
                                                        const u16* __restrict__ Xt,
                                                        const float* __restrict__ bias,
                                                        void* __restrict__ outp, int mode) {
    int b   = blockIdx.z;
    int n0  = blockIdx.x * 64;
    int d0  = blockIdx.y * 64;
    int lane = threadIdx.x & 31, wave = threadIdx.x >> 5;
    int half = lane >> 4, l16 = lane & 15;
    int srow  = wave >> 1;        // 0..3 : 16-row group of dout
    int scol0 = (wave & 1) * 2;   // 0 or 2 : first 16-col group of n

    const u16* wrow = W + (size_t)(d0 + srow * 16 + l16) * D_MODEL;
    const u16* xc0  = Xt + ((size_t)b * NPTS + n0 + scol0 * 16 + l16) * D_MODEL + half * 16;
    const u16* xc1  = xc0 + (size_t)16 * D_MODEL;

    v8f c0, c1;
    #pragma unroll
    for (int r = 0; r < 8; ++r) { c0[r] = 0.0f; c1[r] = 0.0f; }

    #pragma unroll
    for (int k0 = 0; k0 < D_MODEL; k0 += 32) {
        Frag a, b0f, b1f;
        a.q[0]   = *(const uint4*)(wrow + k0 + half * 16);
        a.q[1]   = *(const uint4*)(wrow + k0 + half * 16 + 8);
        b0f.q[0] = *(const uint4*)(xc0 + k0);
        b0f.q[1] = *(const uint4*)(xc0 + k0 + 8);
        b1f.q[0] = *(const uint4*)(xc1 + k0);
        b1f.q[1] = *(const uint4*)(xc1 + k0 + 8);
        c0 = __builtin_amdgcn_wmma_f32_16x16x32_bf16(false, a.v, false, b0f.v, (short)0, c0, false, false);
        c1 = __builtin_amdgcn_wmma_f32_16x16x32_bf16(false, a.v, false, b1f.v, (short)0, c1, false, false);
    }

    // epilogue: C layout lane holds (row r + 8*half, col l16)
    #pragma unroll
    for (int r = 0; r < 8; ++r) {
        int drow = d0 + srow * 16 + r + half * 8;
        float bv = bias[drow];
        float v0 = c0[r] + bv;
        float v1 = c1[r] + bv;
        int nA = n0 + scol0 * 16 + l16;
        int nB = nA + 16;
        if (mode == 2) {
            float* o = (float*)outp;
            o[((size_t)b * D_MODEL + drow) * NPTS + nA] = v0;
            o[((size_t)b * D_MODEL + drow) * NPTS + nB] = v1;
        } else {
            int h  = drow & (HEADS - 1);
            int hd = drow >> 2;
            u16* o = (u16*)outp;
            size_t bh = (size_t)b * HEADS + h;
            if (mode == 0) {
                o[(bh * NPTS + nA) * HDIM + hd] = f2bf(v0);
                o[(bh * NPTS + nB) * HDIM + hd] = f2bf(v1);
            } else {
                size_t rb = (bh * HDIM + hd) * NPTS;
                o[rb + nA] = f2bf(v0);
                o[rb + nB] = f2bf(v1);
            }
        }
    }
}

// ---------------------------------------------------------------------------
// Flash attention with distance modulation.
// Block 256 = 8 waves; each wave owns 16 query rows; m loop in steps of 32.
// K/V tiles streamed into LDS by the Tensor Data Mover (double-buffered),
// wave 0 issues, all waves sync on s_wait_tensorcnt + barrier.
// q/k layout [b,h,n,hd]; v layout [b,h,hd,m]; dm f32 [b,n,m] (pre-scaled).
// Output written transposed: xT[b][n][d] bf16, d = hd*H + h.
// ---------------------------------------------------------------------------
#define PS 40  // P bounce row stride (bf16 elems)

__global__ __launch_bounds__(256) void attn_kernel(const u16* __restrict__ qb,
                                                   const u16* __restrict__ kb,
                                                   const u16* __restrict__ vb,
                                                   const float* __restrict__ dm,
                                                   u16* __restrict__ xt) {
    __shared__ u16 P[8 * 16 * PS];
#if USE_TDM
    __shared__ u16 Kl[2][32 * HDIM];   // [buf][m-row][hd]  4KB each
    __shared__ u16 Vl[2][HDIM * 32];   // [buf][hd-row][m]  4KB each
#endif

    int b = blockIdx.z, h = blockIdx.y;
    int lane = threadIdx.x & 31, wave = threadIdx.x >> 5;
    int half = lane >> 4, l16 = lane & 15;
    int nbase = blockIdx.x * 128 + wave * 16;
    size_t bh = (size_t)b * HEADS + h;
    u16* Pw = P + (size_t)wave * 16 * PS;

    // Q A-fragments (K = hd 0..31 and 32..63)
    Frag aq0, aq1;
    {
        const u16* qrow = qb + (bh * NPTS + nbase + l16) * HDIM;
        aq0.q[0] = *(const uint4*)(qrow + half * 16);
        aq0.q[1] = *(const uint4*)(qrow + half * 16 + 8);
        aq1.q[0] = *(const uint4*)(qrow + 32 + half * 16);
        aq1.q[1] = *(const uint4*)(qrow + 40 + half * 16);
    }

    float mrow[8], lrow[8];
    v8f acc[4];
    #pragma unroll
    for (int r = 0; r < 8; ++r) {
        mrow[r] = -3.0e38f;
        lrow[r] = 0.0f;
        #pragma unroll
        for (int t = 0; t < 4; ++t) acc[t][r] = 0.0f;
    }

    const float* dmb = dm + (size_t)b * NPTS * NPTS;

#if USE_TDM
    if (wave == 0) {  // prologue: fetch first K/V tiles into buffer 0
        tdm_load_2d_bf16((unsigned)(size_t)(void*)&Kl[0][0],
                         kb + bh * NPTS * HDIM, HDIM, 32, HDIM);
        tdm_load_2d_bf16((unsigned)(size_t)(void*)&Vl[0][0],
                         vb + bh * HDIM * NPTS, 32, HDIM, NPTS);
    }
#endif

    for (int m0 = 0; m0 < NPTS; m0 += 32) {
#if USE_TDM
        int cur = (m0 >> 5) & 1;
        if (wave == 0) __builtin_amdgcn_s_wait_tensorcnt(0);
        __syncthreads();  // tiles[cur] ready; all waves done with tiles[1-cur]
        if (wave == 0 && m0 + 32 < NPTS) {
            tdm_load_2d_bf16((unsigned)(size_t)(void*)&Kl[1 - cur][0],
                             kb + (bh * NPTS + m0 + 32) * HDIM, HDIM, 32, HDIM);
            tdm_load_2d_bf16((unsigned)(size_t)(void*)&Vl[1 - cur][0],
                             vb + bh * HDIM * NPTS + m0 + 32, 32, HDIM, NPTS);
        }
        const u16* Kt = &Kl[cur][0];
        const u16* Vt = &Vl[cur][0];
#endif
        // scores: s[mt] = Q(16x64) . K(64x16) for the two 16-col m tiles
        v8f s0, s1;
        #pragma unroll
        for (int r = 0; r < 8; ++r) { s0[r] = 0.0f; s1[r] = 0.0f; }
        {
            Frag bk;
#if USE_TDM
            const u16* kr0 = Kt + (size_t)l16 * HDIM + half * 16;
            const u16* kr1 = Kt + (size_t)(16 + l16) * HDIM + half * 16;
#else
            const u16* kr0 = kb + (bh * NPTS + m0 + l16) * HDIM + half * 16;
            const u16* kr1 = kb + (bh * NPTS + m0 + 16 + l16) * HDIM + half * 16;
#endif
            bk.q[0] = *(const uint4*)(kr0);      bk.q[1] = *(const uint4*)(kr0 + 8);
            s0 = __builtin_amdgcn_wmma_f32_16x16x32_bf16(false, aq0.v, false, bk.v, (short)0, s0, false, false);
            bk.q[0] = *(const uint4*)(kr0 + 32); bk.q[1] = *(const uint4*)(kr0 + 40);
            s0 = __builtin_amdgcn_wmma_f32_16x16x32_bf16(false, aq1.v, false, bk.v, (short)0, s0, false, false);
            bk.q[0] = *(const uint4*)(kr1);      bk.q[1] = *(const uint4*)(kr1 + 8);
            s1 = __builtin_amdgcn_wmma_f32_16x16x32_bf16(false, aq0.v, false, bk.v, (short)0, s1, false, false);
            bk.q[0] = *(const uint4*)(kr1 + 32); bk.q[1] = *(const uint4*)(kr1 + 40);
            s1 = __builtin_amdgcn_wmma_f32_16x16x32_bf16(false, aq1.v, false, bk.v, (short)0, s1, false, false);
        }

        // distance modulation + online softmax (rows: r + 8*half; col = l16)
        #pragma unroll
        for (int r = 0; r < 8; ++r) {
            int row = nbase + r + half * 8;
            const float* dmr = dmb + (size_t)row * NPTS + m0;
            float x0 = s0[r] * dmr[l16];
            float x1 = s1[r] * dmr[16 + l16];

            float t = fmaxf(x0, x1);
            t = fmaxf(t, __shfl_xor(t, 1, 32));
            t = fmaxf(t, __shfl_xor(t, 2, 32));
            t = fmaxf(t, __shfl_xor(t, 4, 32));
            t = fmaxf(t, __shfl_xor(t, 8, 32));

            float newm  = fmaxf(mrow[r], t);
            float alpha = __expf(mrow[r] - newm);
            float p0 = __expf(x0 - newm);
            float p1 = __expf(x1 - newm);
            float ps = p0 + p1;
            ps += __shfl_xor(ps, 1, 32);
            ps += __shfl_xor(ps, 2, 32);
            ps += __shfl_xor(ps, 4, 32);
            ps += __shfl_xor(ps, 8, 32);

            lrow[r] = lrow[r] * alpha + ps;
            mrow[r] = newm;
            #pragma unroll
            for (int t4 = 0; t4 < 4; ++t4) acc[t4][r] *= alpha;

            u16* pr = Pw + (size_t)(r + half * 8) * PS;
            pr[l16]      = f2bf(p0);
            pr[16 + l16] = f2bf(p1);
        }
        __syncthreads();

        // P as A-fragment (16n x 32m)
        Frag ap;
        {
            const u16* pp = Pw + (size_t)l16 * PS;
            ap.q[0] = *(const uint4*)(pp + half * 16);
            ap.q[1] = *(const uint4*)(pp + half * 16 + 8);
        }
        // acc[t] += P(16x32) . V^T(32x16) for 4 hd tiles
        #pragma unroll
        for (int t4 = 0; t4 < 4; ++t4) {
            Frag bv;
#if USE_TDM
            const u16* vr = Vt + (size_t)(t4 * 16 + l16) * 32 + half * 16;
#else
            const u16* vr = vb + (bh * HDIM + t4 * 16 + l16) * NPTS + m0 + half * 16;
#endif
            bv.q[0] = *(const uint4*)(vr);
            bv.q[1] = *(const uint4*)(vr + 8);
            acc[t4] = __builtin_amdgcn_wmma_f32_16x16x32_bf16(false, ap.v, false, bv.v, (short)0, acc[t4], false, false);
        }
    }

    // normalize + write xT[b][n][d] (bf16), d = hd*H + h
    #pragma unroll
    for (int r = 0; r < 8; ++r) {
        float inv = 1.0f / lrow[r];
        int row = nbase + r + half * 8;
        u16* orow = xt + ((size_t)b * NPTS + row) * D_MODEL + h;
        #pragma unroll
        for (int t4 = 0; t4 < 4; ++t4) {
            int hd = t4 * 16 + l16;
            orow[hd * HEADS] = f2bf(acc[t4][r] * inv);
        }
    }
}

// ---------------------------------------------------------------------------
extern "C" void kernel_launch(void* const* d_in, const int* in_sizes, int n_in,
                              void* d_out, int out_size, void* d_ws, size_t ws_size,
                              hipStream_t stream) {
    const float* query    = (const float*)d_in[0];
    const float* key      = (const float*)d_in[1];
    const float* value    = (const float*)d_in[2];
    const float* kpts_src = (const float*)d_in[3];
    const float* kpts_dst = (const float*)d_in[4];
    const float* Wq = (const float*)d_in[5];  const float* bq = (const float*)d_in[6];
    const float* Wk = (const float*)d_in[7];  const float* bk = (const float*)d_in[8];
    const float* Wv = (const float*)d_in[9];  const float* bv = (const float*)d_in[10];
    const float* Wm = (const float*)d_in[11]; const float* bm = (const float*)d_in[12];
    const float* proj_dist = (const float*)d_in[13];

    const size_t ACT = (size_t)BATCH * D_MODEL * NPTS;   // 4,194,304 elems
    const size_t WSZ = (size_t)D_MODEL * D_MODEL;        // 65,536 elems
    char* w = (char*)d_ws;
    u16* qin  = (u16*)w;  w += ACT * 2;   // transposed bf16 inputs [b][n][d]
    u16* kin  = (u16*)w;  w += ACT * 2;
    u16* vin  = (u16*)w;  w += ACT * 2;
    u16* wqb  = (u16*)w;  w += WSZ * 2;
    u16* wkb  = (u16*)w;  w += WSZ * 2;
    u16* wvb  = (u16*)w;  w += WSZ * 2;
    u16* wmb  = (u16*)w;  w += WSZ * 2;
    u16* qbuf = (u16*)w;  w += ACT * 2;
    u16* kbuf = (u16*)w;  w += ACT * 2;
    u16* vbuf = (u16*)w;  w += ACT * 2;
    u16* xbuf = (u16*)w;  w += ACT * 2;   // attention output, xT layout [b][n][d]
    float* dmbuf = (float*)w;             // BATCH*NPTS*NPTS f32 = 32 MB

    // 1. transpose-convert activations; convert weights
    dim3 tg(NPTS / 32, D_MODEL / 32, BATCH);
    cvt_tr_kernel<<<tg, 256, 0, stream>>>(query, qin);
    cvt_tr_kernel<<<tg, 256, 0, stream>>>(key,   kin);
    cvt_tr_kernel<<<tg, 256, 0, stream>>>(value, vin);
    cvt_bf16_kernel<<<dim3((unsigned)(WSZ / 256)), 256, 0, stream>>>(Wq, wqb, (int)WSZ);
    cvt_bf16_kernel<<<dim3((unsigned)(WSZ / 256)), 256, 0, stream>>>(Wk, wkb, (int)WSZ);
    cvt_bf16_kernel<<<dim3((unsigned)(WSZ / 256)), 256, 0, stream>>>(Wv, wvb, (int)WSZ);
    cvt_bf16_kernel<<<dim3((unsigned)(WSZ / 256)), 256, 0, stream>>>(Wm, wmb, (int)WSZ);

    // 2. distance modulation matrix (rank gather + 1/sqrt(HD))
    dist_kernel<<<dim3(NPTS, BATCH), 512, 0, stream>>>(kpts_src, kpts_dst, proj_dist, dmbuf);

    // 3. q/k/v projections (WMMA, LDS-free)
    dim3 pg(NPTS / 64, D_MODEL / 64, BATCH);
    proj_gemm_kernel<<<pg, 256, 0, stream>>>(wqb, qin, bq, qbuf, 0);
    proj_gemm_kernel<<<pg, 256, 0, stream>>>(wkb, kin, bk, kbuf, 0);
    proj_gemm_kernel<<<pg, 256, 0, stream>>>(wvb, vin, bv, vbuf, 1);

    // 4. flash attention with distance modulation (WMMA + TDM double buffering)
    attn_kernel<<<dim3(NPTS / 128, HEADS, BATCH), 256, 0, stream>>>(qbuf, kbuf, vbuf, dmbuf, xbuf);

    // 5. output projection -> f32 d_out (WMMA)
    proj_gemm_kernel<<<pg, 256, 0, stream>>>(wmb, xbuf, bm, d_out, 2);
}
